// Frequency_attention_87084756893876
// MI455X (gfx1250) — matrix operand order repaired and checked by reading, not compiled, not run on hardware
//
#include <hip/hip_runtime.h>
#include <hip/hip_bf16.h>

typedef __attribute__((ext_vector_type(16))) _Float16 v16h;
typedef __attribute__((ext_vector_type(8)))  _Float16 v8h;
typedef __attribute__((ext_vector_type(8)))  float    v8f;

#define NB_   16      // batch
#define NC__  128     // input channels
#define NTOK  640     // tokens per batch (5*128)
#define EMB_  256
#define HEADS_ 8
#define DH_   32
#define T_    640
#define BH_   128     // NB_*HEADS_
#define NHASH 8
#define NBUCK 10
#define NCH   80      // chunks = NHASH*NBUCK
#define LTOT  5120    // NHASH*T_
#define LPAD  8192

// ---------------- WMMA fragment loaders (CDNA5 16x16x32 f16 layouts) ----------------

// A matrix 16x32 f16, row-major source with leading dim ld (halfs).
// lane L: M = L%16 ; K in {b..b+7} u {b+16..b+23}, b = (L/16)*8
__device__ inline v16h load_a_frag(const _Float16* __restrict__ tile, int ld, int lane) {
    int m  = lane & 15;
    int kb = (lane >> 4) * 8;
    const _Float16* p = tile + (size_t)m * ld + kb;
    v8h lo = *(const v8h*)(p);
    v8h hi = *(const v8h*)(p + 16);
    v16h a;
    #pragma unroll
    for (int e = 0; e < 8; ++e) { a[e] = lo[e]; a[8 + e] = hi[e]; }
    return a;
}

// B matrix 32x16 f16 where B[k][n] = BT[n][k], BT row-major leading dim ldk (contiguous loads).
// lane L: N = L%16 ; K in {kb..kb+7} u {kb+16..kb+23}, kb = (L/16)*8
__device__ inline v16h load_b_frag_t(const _Float16* __restrict__ BT, int ldk, int lane) {
    int n  = lane & 15;
    int kb = (lane >> 4) * 8;
    const _Float16* p = BT + (size_t)n * ldk + kb;
    v8h lo = *(const v8h*)(p);
    v8h hi = *(const v8h*)(p + 16);
    v16h b;
    #pragma unroll
    for (int e = 0; e < 8; ++e) { b[e] = lo[e]; b[8 + e] = hi[e]; }
    return b;
}

// ---------------- Kernel 0: f32 -> f16 convert + transpose (256x256) ----------------
__global__ void k_cvt_t(const float* __restrict__ a, _Float16* __restrict__ ot) {
    int i = blockIdx.x * blockDim.x + threadIdx.x;   // i = k*256 + n
    if (i < EMB_ * EMB_) {
        int k = i >> 8, n = i & 255;
        ot[(size_t)n * EMB_ + k] = (_Float16)a[i];
    }
}

// ---------------- Kernel 1: maxpool + Haar DWT -> tokens f16 (10240 x 256) ----------------
__global__ void k_pre(const float* __restrict__ x, _Float16* __restrict__ tokens) {
    int bc = blockIdx.x;                 // 0..2047
    int b = bc / NC__, c = bc % NC__;
    int s = threadIdx.x;                 // 0..255 : spatial position in 16x16
    int h = s >> 4, w = s & 15;
    const float* xp = x + (size_t)(b * NC__ + c) * 32 * 32;

    // maxpool 3x3 stride2 pad1 (pad value -inf)
    float m = -INFINITY;
    int r0 = 2 * h - 1, c0 = 2 * w - 1;
    #pragma unroll
    for (int dr = 0; dr < 3; ++dr) {
        int r = r0 + dr; if (r < 0 || r >= 32) continue;
        #pragma unroll
        for (int dc = 0; dc < 3; ++dc) {
            int cc = c0 + dc; if (cc < 0 || cc >= 32) continue;
            m = fmaxf(m, xp[r * 32 + cc]);
        }
    }
    // Haar DWT (2x2 blocks)
    float a  = xp[(2 * h) * 32 + 2 * w];
    float bb = xp[(2 * h) * 32 + 2 * w + 1];
    float cv = xp[(2 * h + 1) * 32 + 2 * w];
    float d  = xp[(2 * h + 1) * 32 + 2 * w + 1];
    float ll = (a + bb + cv + d) * 0.5f;
    float lh = (a - bb + cv - d) * 0.5f;
    float hl = (a + bb - cv - d) * 0.5f;
    float hh = (a - bb - cv + d) * 0.5f;

    size_t rb = (size_t)b * NTOK;
    tokens[(rb + c) * EMB_ + s]       = (_Float16)m;
    tokens[(rb + 128 + c) * EMB_ + s] = (_Float16)ll;
    tokens[(rb + 256 + c) * EMB_ + s] = (_Float16)lh;
    tokens[(rb + 384 + c) * EMB_ + s] = (_Float16)hl;
    tokens[(rb + 512 + c) * EMB_ + s] = (_Float16)hh;
}

// ---------------- Kernel 2: WMMA GEMM  C[M x 256] = A[M x 256] * W[256 x 256] ----------------
// WT is the transposed f16 weight (WT[n][k] = W[k][n]) -> contiguous B fragments.
// One block = one 16-row strip; wave w covers columns [w*32, w*32+32) with two
// accumulators sharing one A fragment per K-step (2 WMMAs per A load).
// mode 0: write qk f32 to (bh, t, dh) layout
// mode 1: write v  f16 to (bh, t, dh) layout
// mode 2: write f32 out row-major + bias
__global__ void k_gemm(const _Float16* __restrict__ A, const _Float16* __restrict__ WT,
                       int Mtiles, int mode,
                       float* __restrict__ out_qk, _Float16* __restrict__ out_v,
                       float* __restrict__ out_final, const float* __restrict__ bias) {
    int lane = threadIdx.x & 31;
    int wave = threadIdx.x >> 5;         // 0..7 -> column pair
    int tm = blockIdx.x;                 // row tile
    if (tm >= Mtiles) return;

    v8f acc0 = {};
    v8f acc1 = {};
    const _Float16* Abase  = A  + (size_t)tm * 16 * EMB_;
    const _Float16* Bbase0 = WT + (size_t)(wave * 32) * EMB_;
    const _Float16* Bbase1 = Bbase0 + (size_t)16 * EMB_;
    #pragma unroll
    for (int kk = 0; kk < EMB_; kk += 32) {
        if (kk + 32 < EMB_)
            __builtin_prefetch(Abase + ((size_t)(lane & 15)) * EMB_ + kk + 32, 0, 3);
        v16h af  = load_a_frag(Abase + kk, EMB_, lane);
        v16h bf0 = load_b_frag_t(Bbase0 + kk, EMB_, lane);
        v16h bf1 = load_b_frag_t(Bbase1 + kk, EMB_, lane);
        acc0 = __builtin_amdgcn_wmma_f32_16x16x32_f16(false, af, false, bf0, (short)0, acc0, false, false);
        acc1 = __builtin_amdgcn_wmma_f32_16x16x32_f16(false, af, false, bf1, (short)0, acc1, false, false);
    }
    int hf = lane >> 4, nl = lane & 15;
    #pragma unroll
    for (int half = 0; half < 2; ++half) {
        #pragma unroll
        for (int g = 0; g < 8; ++g) {
            int row = tm * 16 + g + 8 * hf;
            int col = wave * 32 + half * 16 + nl;
            float v = half ? acc1[g] : acc0[g];
            if (mode == 0) {
                int b = row / NTOK, n = row % NTOK;
                int hh = col >> 5, e = col & 31;
                out_qk[((size_t)(b * HEADS_ + hh) * T_ + n) * DH_ + e] = v;
            } else if (mode == 1) {
                int b = row / NTOK, n = row % NTOK;
                int hh = col >> 5, e = col & 31;
                out_v[((size_t)(b * HEADS_ + hh) * T_ + n) * DH_ + e] = (_Float16)v;
            } else {
                out_final[(size_t)row * EMB_ + col] = v + bias[col];
            }
        }
    }
}

// ---------------- Kernel 3: LSH hashing -> packed sort keys ----------------
__global__ void k_hash(const float* __restrict__ qk, const float* __restrict__ rot,
                       unsigned int* __restrict__ keys) {
    int bh = blockIdx.x; // 128
    for (int t = threadIdx.x; t < T_; t += blockDim.x) {
        float q[DH_];
        const float* qp = qk + ((size_t)bh * T_ + t) * DH_;
        #pragma unroll
        for (int k = 0; k < DH_; ++k) q[k] = qp[k];
        for (int h = 0; h < NHASH; ++h) {
            float r[5];
            #pragma unroll
            for (int i = 0; i < 5; ++i) {
                float s = 0.f;
                #pragma unroll
                for (int k = 0; k < DH_; ++k) s += q[k] * rot[(k * NHASH + h) * 5 + i];
                r[i] = s;
            }
            int best = 0; float bv = r[0];
            #pragma unroll
            for (int i = 1; i < 10; ++i) {
                float vv = (i < 5) ? r[i] : -r[i - 5];
                if (vv > bv) { bv = vv; best = i; }
            }
            int bucket = best + h * NBUCK;                       // [0,80)
            unsigned int bat = (unsigned)(T_ * bucket + t);      // < 2^16
            unsigned int p   = (unsigned)(h * T_ + t);           // < 2^13
            keys[(size_t)bh * LTOT + p] = (bat << 13) | p;       // stable-sort key
        }
    }
}

// ---------------- Kernel 4: per-bh bitonic sort (LDS) -> st, undo ----------------
__global__ void k_sort(const unsigned int* __restrict__ keys,
                       int* __restrict__ st, int* __restrict__ undo) {
    __shared__ unsigned int sk[LPAD];
    int bh = blockIdx.x;
    int tid = threadIdx.x; // 512
    for (int i = tid; i < LPAD; i += 512)
        sk[i] = (i < LTOT) ? keys[(size_t)bh * LTOT + i] : 0xFFFFFFFFu;
    __syncthreads();
    for (int k = 2; k <= LPAD; k <<= 1) {
        for (int j = k >> 1; j > 0; j >>= 1) {
            for (int i = tid; i < LPAD; i += 512) {
                int ixj = i ^ j;
                if (ixj > i) {
                    unsigned int a = sk[i], b = sk[ixj];
                    bool up = ((i & k) == 0);
                    if ((a > b) == up) { sk[i] = b; sk[ixj] = a; }
                }
            }
            __syncthreads();
        }
    }
    for (int j = tid; j < LTOT; j += 512) {
        int p = (int)(sk[j] & 0x1FFFu);          // original flat position
        st[(size_t)bh * LTOT + j] = p % T_;      // token index at sorted slot j
        undo[(size_t)bh * LTOT + p] = j;         // inverse permutation
    }
}

// ---------------- Kernel 5: chunked LSH attention (WMMA) ----------------
__global__ void k_attn(const float* __restrict__ qk, const _Float16* __restrict__ v,
                       const int* __restrict__ st,
                       _Float16* __restrict__ so, float* __restrict__ slog) {
    __shared__ __align__(32) _Float16 s_bq[64 * DH_];        // 4 KB  [i][k]
    __shared__ __align__(32) _Float16 s_bk[128 * DH_];       // 8 KB  [j][k]
    __shared__ __align__(32) _Float16 s_bvT[DH_ * 128];      // 8 KB  [e][j]  (transposed V)
    __shared__ __align__(32) float    s_dots[64 * 128];      // 32 KB (reused for f16 probs)
    __shared__ int   s_qt[64];
    __shared__ int   s_kt[128];
    __shared__ float s_norm[128];
    __shared__ float s_lse[64];

    int chunk = blockIdx.x, bh = blockIdx.y;
    int tid = threadIdx.x;               // 256 threads = 8 waves
    int lane = tid & 31, wave = tid >> 5;

    const int* stp = st + (size_t)bh * LTOT;
    if (tid < 64) s_qt[tid] = stp[chunk * 64 + tid];
    if (tid < 128) {
        int src = (tid < 64) ? chunk : (chunk + NCH - 1) % NCH;  // look-one-back
        s_kt[tid] = stp[src * 64 + (tid & 63)];
    }
    __syncthreads();

    const float*    qkb = qk + (size_t)bh * T_ * DH_;
    const _Float16* vb  = v  + (size_t)bh * T_ * DH_;

    for (int idx = tid; idx < 64 * DH_; idx += 256) {
        int i = idx >> 5, k = idx & 31;
        s_bq[idx] = (_Float16)qkb[(size_t)s_qt[i] * DH_ + k];
    }
    if (tid < 128) {
        const float* r = qkb + (size_t)s_kt[tid] * DH_;
        float s2 = 0.f;
        #pragma unroll
        for (int k = 0; k < DH_; ++k) { float vv = r[k]; s2 += vv * vv; }
        s_norm[tid] = fmaxf(sqrtf(s2), 1e-6f);
    }
    __syncthreads();
    for (int idx = tid; idx < 128 * DH_; idx += 256) {
        int j = idx >> 5, k = idx & 31;
        float inv = 1.f / s_norm[j];
        s_bk[idx] = (_Float16)(qkb[(size_t)s_kt[j] * DH_ + k] * inv);
        s_bvT[k * 128 + j] = vb[(size_t)s_kt[j] * DH_ + k];   // transposed store
    }
    __syncthreads();

    // dots = bq (64x32) @ bk^T (32x128) : 32 tiles of 16x16, 4 per wave, K=32 (one WMMA each)
    const float scale = 0.17677669529663687f; // 32^-0.5
    #pragma unroll
    for (int q = 0; q < 4; ++q) {
        int tt = wave * 4 + q;
        int tm = tt >> 3, tn = tt & 7;
        v16h af = load_a_frag(s_bq + tm * 16 * DH_, DH_, lane);
        v16h bf = load_b_frag_t(s_bk + tn * 16 * DH_, DH_, lane);
        v8f acc = {};
        acc = __builtin_amdgcn_wmma_f32_16x16x32_f16(false, af, false, bf, (short)0, acc, false, false);
        int hf = lane >> 4, nl = lane & 15;
        int jc = tn * 16 + nl;
        #pragma unroll
        for (int g = 0; g < 8; ++g) {
            int ir = tm * 16 + g + 8 * hf;
            float vv = acc[g] * scale;
            if (s_qt[ir] == s_kt[jc]) vv = -50000.0f;   // self-attention mask
            s_dots[ir * 128 + jc] = vv;
        }
    }
    __syncthreads();

    // row softmax with logsumexp (64 rows)
    if (tid < 64) {
        float mx = -INFINITY;
        for (int j = 0; j < 128; ++j) mx = fmaxf(mx, s_dots[tid * 128 + j]);
        float s = 0.f;
        for (int j = 0; j < 128; ++j) s += expf(s_dots[tid * 128 + j] - mx);
        float lse = mx + logf(s);
        s_lse[tid] = lse;
        slog[(size_t)bh * LTOT + chunk * 64 + tid] = lse;
    }
    __syncthreads();

    // probs (f16) overlaid on the dots buffer
    float pv[32];
    #pragma unroll
    for (int r = 0; r < 32; ++r) {
        int idx = tid + r * 256;
        pv[r] = expf(s_dots[idx] - s_lse[idx >> 7]);
    }
    __syncthreads();
    _Float16* s_probs = (_Float16*)s_dots;
    #pragma unroll
    for (int r = 0; r < 32; ++r) {
        int idx = tid + r * 256;
        s_probs[idx] = (_Float16)pv[r];
    }
    __syncthreads();

    // bo = probs (64x128) @ bv (128x32) : 8 tiles, one per wave, K=128
    // B fragments from transposed V: B[k][n] = bv[k][n] = s_bvT[n][k] -> contiguous
    {
        int tm = wave >> 1, tn = wave & 1;
        v8f acc = {};
        #pragma unroll
        for (int kk = 0; kk < 128; kk += 32) {
            v16h af = load_a_frag(s_probs + tm * 16 * 128 + kk, 128, lane);
            v16h bf = load_b_frag_t(s_bvT + (size_t)(tn * 16) * 128 + kk, 128, lane);
            acc = __builtin_amdgcn_wmma_f32_16x16x32_f16(false, af, false, bf, (short)0, acc, false, false);
        }
        int hf = lane >> 4, nl = lane & 15;
        #pragma unroll
        for (int g = 0; g < 8; ++g) {
            int ir = tm * 16 + g + 8 * hf;
            int ec = tn * 16 + nl;
            so[((size_t)bh * LTOT + chunk * 64 + ir) * DH_ + ec] = (_Float16)acc[g];
        }
    }
}

// ---------------- Kernel 6: unsort + cross-hash logsumexp combine ----------------
__global__ void k_combine(const _Float16* __restrict__ so, const float* __restrict__ slog,
                          const int* __restrict__ undo, _Float16* __restrict__ ofin) {
    int gw = (blockIdx.x * blockDim.x + threadIdx.x) >> 5;  // one wave per (bh, t)
    int lane = threadIdx.x & 31;
    if (gw >= BH_ * T_) return;
    int bh = gw / T_, t = gw % T_;
    const int*   up = undo + (size_t)bh * LTOT;
    const float* lp = slog + (size_t)bh * LTOT;

    float lg[NHASH]; int jj[NHASH];
    float mx = -INFINITY;
    #pragma unroll
    for (int h = 0; h < NHASH; ++h) {
        int j = up[h * T_ + t];
        jj[h] = j; lg[h] = lp[j];
        mx = fmaxf(mx, lg[h]);
    }
    float s = 0.f;
    #pragma unroll
    for (int h = 0; h < NHASH; ++h) s += expf(lg[h] - mx);
    float lse = mx + logf(s);

    float acc = 0.f;
    #pragma unroll
    for (int h = 0; h < NHASH; ++h) {
        float w = expf(lg[h] - lse);
        acc += w * (float)so[((size_t)bh * LTOT + jj[h]) * DH_ + lane];
    }
    int b = bh >> 3, hd = bh & 7;
    ofin[((size_t)b * T_ + t) * EMB_ + hd * DH_ + lane] = (_Float16)acc;
}

// ---------------- Host launcher ----------------
extern "C" void kernel_launch(void* const* d_in, const int* in_sizes, int n_in,
                              void* d_out, int out_size, void* d_ws, size_t ws_size,
                              hipStream_t stream) {
    const float* x     = (const float*)d_in[0];
    const float* w_qk  = (const float*)d_in[1];
    const float* w_v   = (const float*)d_in[2];
    const float* w_out = (const float*)d_in[3];
    const float* b_out = (const float*)d_in[4];
    const float* rot   = (const float*)d_in[5];
    float* out = (float*)d_out;

    char* ws = (char*)d_ws;
    size_t off = 0;
    auto alloc = [&](size_t bytes) {
        void* p = ws + off;
        off = (off + bytes + 255) & ~(size_t)255;
        return p;
    };
    const size_t M = (size_t)NB_ * NTOK;                 // 10240 tokens
    _Float16* tokens = (_Float16*)alloc(M * EMB_ * 2);
    _Float16* wqk_t  = (_Float16*)alloc(EMB_ * EMB_ * 2);
    _Float16* wv_t   = (_Float16*)alloc(EMB_ * EMB_ * 2);
    _Float16* wout_t = (_Float16*)alloc(EMB_ * EMB_ * 2);
    float*    qkf    = (float*)   alloc((size_t)BH_ * T_ * DH_ * 4);
    _Float16* vh     = (_Float16*)alloc((size_t)BH_ * T_ * DH_ * 2);
    unsigned int* keys = (unsigned int*)alloc((size_t)BH_ * LTOT * 4);
    int*      stb    = (int*)     alloc((size_t)BH_ * LTOT * 4);
    int*      undob  = (int*)     alloc((size_t)BH_ * LTOT * 4);
    float*    slogb  = (float*)   alloc((size_t)BH_ * LTOT * 4);
    _Float16* sob    = (_Float16*)alloc((size_t)BH_ * LTOT * DH_ * 2);
    _Float16* ofin   = (_Float16*)alloc(M * EMB_ * 2);

    // weight conversion + transpose (WT[n][k])
    k_cvt_t<<<(EMB_ * EMB_ + 255) / 256, 256, 0, stream>>>(w_qk,  wqk_t);
    k_cvt_t<<<(EMB_ * EMB_ + 255) / 256, 256, 0, stream>>>(w_v,   wv_t);
    k_cvt_t<<<(EMB_ * EMB_ + 255) / 256, 256, 0, stream>>>(w_out, wout_t);

    // preprocess: maxpool + haar -> tokens
    k_pre<<<NB_ * NC__, 256, 0, stream>>>(x, tokens);

    // projections (WMMA GEMMs): one block per 16-row strip
    const int Mtiles = (int)(M / 16);    // 640
    k_gemm<<<Mtiles, 256, 0, stream>>>(tokens, wqk_t, Mtiles, 0, qkf, nullptr, nullptr, nullptr);
    k_gemm<<<Mtiles, 256, 0, stream>>>(tokens, wv_t,  Mtiles, 1, nullptr, vh, nullptr, nullptr);

    // LSH hash keys, per-bh bitonic sort
    k_hash<<<BH_, 256, 0, stream>>>(qkf, rot, keys);
    k_sort<<<BH_, 512, 0, stream>>>(keys, stb, undob);

    // chunked attention
    k_attn<<<dim3(NCH, BH_), 256, 0, stream>>>(qkf, vh, stb, sob, slogb);

    // unsort + combine across hashes
    k_combine<<<(BH_ * T_ * 32) / 256, 256, 0, stream>>>(sob, slogb, undob, ofin);

    // output projection + bias (WMMA GEMM)
    k_gemm<<<Mtiles, 256, 0, stream>>>(ofin, wout_t, Mtiles, 2, nullptr, nullptr, out, b_out);
}
